// EAMP_70866960384260
// MI455X (gfx1250) — matrix operand order repaired and checked by reading, not compiled, not run on hardware
//
#include <hip/hip_runtime.h>
#include <hip/hip_bf16.h>
#include <math.h>

typedef __attribute__((ext_vector_type(16))) _Float16 v16h;
typedef __attribute__((ext_vector_type(8)))  _Float16 v8h;
typedef __attribute__((ext_vector_type(8)))  float    v8f;

#define N_NODES 10000
#define N_EDGES 160000
#define AGG_CH  576   // 224 (agg0) + 192 (agg1 u*3+m) + 160 (agg2 u*5+m)
#define INV_NEIGH 0.25f
#define EPS 1e-8f

// 64 edges / block = four 16-row M tiles
#define EDGES_PER_BLK 64
#define HA0  (EDGES_PER_BLK * 672)
#define HAP  (EDGES_PER_BLK * 288)
#define HAN  (EDGES_PER_BLK * 288)
#define HP2  (EDGES_PER_BLK * 96)
#define HN2  (EDGES_PER_BLK * 96)
// halves + red[256] + 4 stat arrays[16] + src/dst[64]
#define EDGE_SMEM_BYTES ((HA0 + HAP + HAN + HP2 + HN2) * 2 + 256 * 4 + 4 * 16 * 4 + 2 * 64 * 4)

static __device__ __forceinline__ float sigmoidf_(float x) { return 1.0f / (1.0f + __expf(-x)); }

static __device__ __forceinline__ v8f wmma16(v16h a, v16h b, v8f c) {
    return __builtin_amdgcn_wmma_f32_16x16x32_f16(false, a, false, b, (short)0, c, false, false);
}

// A fragment: rows stored K-major in LDS. lanes 0-15: K [kb,kb+8)+[kb+16,kb+24); lanes 16-31: +8
static __device__ __forceinline__ v16h load_a16(const _Float16* rowp, int kb, int hi) {
    v8h a = *(const v8h*)(rowp + kb + hi * 8);
    v8h b = *(const v8h*)(rowp + kb + 16 + hi * 8);
    return __builtin_shufflevector(a, b, 0,1,2,3,4,5,6,7,8,9,10,11,12,13,14,15);
}

// B fragment: packed so each lane's 16 halves are contiguous
static __device__ __forceinline__ v16h load_b16(const _Float16* Wp, int frag, int lane) {
    const v8h* p = (const v8h*)(Wp + (((size_t)frag * 32 + lane) << 4));
    v8h a = p[0];
    v8h b = p[1];
    return __builtin_shufflevector(a, b, 0,1,2,3,4,5,6,7,8,9,10,11,12,13,14,15);
}

static __device__ __forceinline__ float wave_sum(float x) {
    #pragma unroll
    for (int m = 16; m >= 1; m >>= 1) x += __shfl_xor(x, m, 32);
    return x;
}

// ---------------------------------------------------------------------------
// Weight pack: f32 (K,N) row-major -> f16 WMMA B fragments, each lane 16 halves
// fragment order [ks][nc][lane][j], k = ks*32 + (j>=8)*16 + (lane>=16)*8 + j%8, n = nc*16 + lane%16
// ---------------------------------------------------------------------------
__global__ __launch_bounds__(256) void pack_w_kernel(const float* __restrict__ W,
                                                     _Float16* __restrict__ out,
                                                     int K, int N, float scale) {
    int p = blockIdx.x * 256 + threadIdx.x;
    if (p >= K * N) return;
    int j    = p & 15;
    int lane = (p >> 4) & 31;
    int fn   = p >> 9;
    int NC   = N >> 4;
    int nc   = fn % NC;
    int ks   = fn / NC;
    int k = ks * 32 + (j >> 3) * 16 + (lane >> 4) * 8 + (j & 7);
    int n = nc * 16 + (lane & 15);
    out[p] = (_Float16)(W[k * N + n] * scale);
}

// ---------------------------------------------------------------------------
// Node separable LayerNorm -> f16 irreps [x0(128) | x1 u*3+m (192) | x2 u*5+m (160)]
// one wave per node
// ---------------------------------------------------------------------------
__global__ __launch_bounds__(256) void node_ln_kernel(const float* __restrict__ nf,
                                                      const float* __restrict__ gamma,
                                                      const float* __restrict__ beta,
                                                      _Float16* __restrict__ xnh) {
    int lane = threadIdx.x & 31;
    int node = blockIdx.x * 8 + (threadIdx.x >> 5);
    const float* x = nf + (size_t)node * 480;
    float v[15];
    #pragma unroll
    for (int j = 0; j < 15; ++j) v[j] = x[lane + 32 * j];
    float s4 = v[0] + v[1] + v[2] + v[3];
    float mu = wave_sum(s4) * (1.0f / 128.0f);
    float q = 0.f;
    #pragma unroll
    for (int j = 0; j < 4; ++j) { float d = v[j] - mu; q += d * d; }
    float rstd = 1.0f / sqrtf(wave_sum(q) * (1.0f / 128.0f) + EPS);
    float q1 = 0.f, q2 = 0.f;
    #pragma unroll
    for (int j = 4; j < 10; ++j) q1 += v[j] * v[j];
    #pragma unroll
    for (int j = 10; j < 15; ++j) q2 += v[j] * v[j];
    float m1 = wave_sum(q1) * (1.0f / 192.0f);
    float m2 = wave_sum(q2) * (1.0f / 160.0f);
    float rn = 1.0f / sqrtf(0.5f * (m1 + m2) + EPS);
    _Float16* o = xnh + (size_t)node * 480;
    #pragma unroll
    for (int j = 0; j < 15; ++j) {
        int c = lane + 32 * j;
        float r;
        if (c < 128)       r = gamma[c] * (v[j] - mu) * rstd + beta[c];
        else if (c < 320)  { int u = (c - 128) / 3; r = v[j] * rn * gamma[128 + u]; }
        else               { int u = (c - 320) / 5; r = v[j] * rn * gamma[192 + u]; }
        o[c] = (_Float16)r;
    }
}

// scatter a normalized element (source s: 0=src-node,1=edge,2=dst-node, channel c of 480)
// into the assembled A matrices (K-major rows)
static __device__ __forceinline__ void scatter_elem(int r, int s, int c, _Float16 val,
                                                    _Float16* A0, _Float16* Ap1, _Float16* An1,
                                                    _Float16* P2, _Float16* N2) {
    if (c < 128) { A0[r * 672 + s * 128 + c] = val; return; }
    if (c < 320) {
        int cc = c - 128, u = cc / 3, m = cc - 3 * u;
        if (m == 1)      A0 [r * 672 + 384 + s * 64 + u] = val;
        else if (m == 2) Ap1[r * 288 +       s * 64 + u] = val;
        else             An1[r * 288 +       s * 64 + u] = val;
        return;
    }
    int cc = c - 320, u = cc / 5, m = cc - 5 * u;
    switch (m) {
        case 2:  A0 [r * 672 + 576 + s * 32 + u] = val; break;
        case 3:  Ap1[r * 288 + 192 + s * 32 + u] = val; break;
        case 1:  An1[r * 288 + 192 + s * 32 + u] = val; break;
        case 4:  P2 [r * 96  +       s * 32 + u] = val; break;
        default: N2 [r * 96  +       s * 32 + u] = val; break;
    }
}

// scatter one 16x16 f32 accumulator tile: rows = edges (M-tile mt), cols -> agg channel ch
static __device__ __forceinline__ void scatter_acc(v8f acc, const int* s_dst, int mt, int hi,
                                                   int ch, float* __restrict__ agg) {
    #pragma unroll
    for (int vv = 0; vv < 8; ++vv) {
        int m = mt * 16 + vv + hi * 8;
        unsafeAtomicAdd(agg + (size_t)s_dst[m] * AGG_CH + ch, acc[vv] * INV_NEIGH);
    }
}

// ---------------------------------------------------------------------------
// Fused edge kernel: edge-LN + node gather + 5 assembled GEMMs via WMMA +
// register->atomic message scatter.  64 edges / block (four 16-row M tiles),
// 256 threads (8 waves), quad-M x dual-N accumulation + B prefetch.
// ---------------------------------------------------------------------------
__global__ __launch_bounds__(256) void edge_kernel(
        const float* __restrict__ edge_feat,
        const _Float16* __restrict__ xnh,
        const float* __restrict__ geg, const float* __restrict__ geb,
        const int* __restrict__ esrc, const int* __restrict__ edst,
        const _Float16* __restrict__ W0p,
        const _Float16* __restrict__ W1rp, const _Float16* __restrict__ W1ip,
        const _Float16* __restrict__ W1inp,
        const _Float16* __restrict__ W2rp, const _Float16* __restrict__ W2ip,
        const _Float16* __restrict__ W2inp,
        float* __restrict__ agg) {
    extern __shared__ __align__(16) char smem[];
    _Float16* sA0  = (_Float16*)smem;
    _Float16* sAp1 = sA0  + HA0;
    _Float16* sAn1 = sAp1 + HAP;
    _Float16* sP2  = sAn1 + HAN;
    _Float16* sN2  = sP2  + HP2;
    float* red    = (float*)(sN2 + HN2);
    float* s_mu   = red + 256;
    float* s_rstd = s_mu + 16;
    float* s_rn   = s_rstd + 16;
    float* s_tmp  = s_rn + 16;
    int*   s_src  = (int*)(s_tmp + 16);
    int*   s_dst  = s_src + EDGES_PER_BLK;

    const int tid = threadIdx.x;
    const int r = tid >> 4;      // edge row helper group (0..15)
    const int t = tid & 15;      // helper thread within row
    const int e0 = blockIdx.x * EDGES_PER_BLK;

    if (tid < EDGES_PER_BLK) { s_src[tid] = esrc[e0 + tid]; s_dst[tid] = edst[e0 + tid]; }

    // ---- stage 1: edge LN + gather, four 16-edge passes -------------------
    for (int pass = 0; pass < 4; ++pass) {
        const int rr = r + 16 * pass;
        const float* ef = edge_feat + (size_t)(e0 + rr) * 480;
        float v[30];
        #pragma unroll
        for (int j = 0; j < 30; ++j) v[j] = ef[t + 16 * j];

        float p = 0.f;
        #pragma unroll
        for (int j = 0; j < 8; ++j) p += v[j];
        red[tid] = p; __syncthreads();
        if (t == 0) { float s = 0; for (int i = 0; i < 16; ++i) s += red[(r << 4) + i]; s_mu[r] = s * (1.f/128.f); }
        __syncthreads();
        float mu = s_mu[r];
        p = 0.f;
        #pragma unroll
        for (int j = 0; j < 8; ++j) { float d = v[j] - mu; p += d * d; }
        red[tid] = p; __syncthreads();
        if (t == 0) { float s = 0; for (int i = 0; i < 16; ++i) s += red[(r << 4) + i];
                      s_rstd[r] = 1.0f / sqrtf(s * (1.f/128.f) + EPS); }
        __syncthreads();
        p = 0.f;
        #pragma unroll
        for (int j = 8; j < 20; ++j) p += v[j] * v[j];
        red[tid] = p; __syncthreads();
        if (t == 0) { float s = 0; for (int i = 0; i < 16; ++i) s += red[(r << 4) + i]; s_tmp[r] = s * (1.f/192.f); }
        __syncthreads();
        p = 0.f;
        #pragma unroll
        for (int j = 20; j < 30; ++j) p += v[j] * v[j];
        red[tid] = p; __syncthreads();
        if (t == 0) { float s = 0; for (int i = 0; i < 16; ++i) s += red[(r << 4) + i];
                      s_rn[r] = 1.0f / sqrtf(0.5f * (s_tmp[r] + s * (1.f/160.f)) + EPS); }
        __syncthreads();
        float rstd = s_rstd[r], rn = s_rn[r];

        #pragma unroll
        for (int j = 0; j < 30; ++j) {
            int c = t + 16 * j;
            float o;
            if (c < 128)      o = geg[c] * (v[j] - mu) * rstd + geb[c];
            else if (c < 320) { int u = (c - 128) / 3; o = v[j] * rn * geg[128 + u]; }
            else              { int u = (c - 320) / 5; o = v[j] * rn * geg[192 + u]; }
            scatter_elem(rr, 1, c, (_Float16)o, sA0, sAp1, sAn1, sP2, sN2);
        }
        const _Float16* xs = xnh + (size_t)s_src[rr] * 480;
        const _Float16* xd = xnh + (size_t)s_dst[rr] * 480;
        #pragma unroll
        for (int j = 0; j < 30; ++j) {
            int c = t + 16 * j;
            scatter_elem(rr, 0, c, xs[c], sA0, sAp1, sAn1, sP2, sN2);
            scatter_elem(rr, 2, c, xd[c], sA0, sAp1, sAn1, sP2, sN2);
        }
        __syncthreads();
    }

    // ---- stage 2: WMMA GEMMs + scatter -----------------------------------
    const int wave = tid >> 5;
    const int lane = tid & 31;
    const int mrow = lane & 15;
    const int hi   = lane >> 4;
    const int n    = lane & 15;

    const _Float16* Ar[4];
    #pragma unroll
    for (int mt = 0; mt < 4; ++mt) Ar[mt] = sA0 + (mt * 16 + mrow) * 672;

    // ---- y = in0 @ W0 : K=672 (21 ksteps), N=320 (20 nchunks) ------------
    // main: dual-N (nc0 = wave, nc1 = wave+8) x quad-M, B prefetched
    {
        const int nc0 = wave, nc1 = wave + 8;
        v8f A0[4] = {}, A1[4] = {};
        v16h b0 = load_b16(W0p, 0 * 20 + nc0, lane);
        v16h b1 = load_b16(W0p, 0 * 20 + nc1, lane);
        for (int ks = 0; ks < 21; ++ks) {
            v16h f[4];
            #pragma unroll
            for (int mt = 0; mt < 4; ++mt) f[mt] = load_a16(Ar[mt], ks * 32, hi);
            v16h b0n = b0, b1n = b1;
            if (ks < 20) {
                b0n = load_b16(W0p, (ks + 1) * 20 + nc0, lane);
                b1n = load_b16(W0p, (ks + 1) * 20 + nc1, lane);
            }
            #pragma unroll
            for (int mt = 0; mt < 4; ++mt) A0[mt] = wmma16(f[mt], b0, A0[mt]);
            #pragma unroll
            for (int mt = 0; mt < 4; ++mt) A1[mt] = wmma16(f[mt], b1, A1[mt]);
            b0 = b0n; b1 = b1n;
        }
        int cy0 = nc0 * 16 + n, cy1 = nc1 * 16 + n;
        int ch0 = (cy0 < 224) ? cy0 : (cy0 < 288) ? 224 + (cy0 - 224) * 3 + 1 : 416 + (cy0 - 288) * 5 + 2;
        int ch1 = (cy1 < 224) ? cy1 : (cy1 < 288) ? 224 + (cy1 - 224) * 3 + 1 : 416 + (cy1 - 288) * 5 + 2;
        #pragma unroll
        for (int mt = 0; mt < 4; ++mt) {
            scatter_acc(A0[mt], s_dst, mt, hi, ch0, agg);
            scatter_acc(A1[mt], s_dst, mt, hi, ch1, agg);
        }
    }
    // leftover chunks 16..19: waves 0..3, quad-M
    if (wave < 4) {
        const int nc = 16 + wave;
        v8f A[4] = {};
        v16h b = load_b16(W0p, 0 * 20 + nc, lane);
        for (int ks = 0; ks < 21; ++ks) {
            v16h f[4];
            #pragma unroll
            for (int mt = 0; mt < 4; ++mt) f[mt] = load_a16(Ar[mt], ks * 32, hi);
            v16h bn = b;
            if (ks < 20) bn = load_b16(W0p, (ks + 1) * 20 + nc, lane);
            #pragma unroll
            for (int mt = 0; mt < 4; ++mt) A[mt] = wmma16(f[mt], b, A[mt]);
            b = bn;
        }
        int cy = nc * 16 + n;
        int ch = (cy < 288) ? 224 + (cy - 224) * 3 + 1 : 416 + (cy - 288) * 5 + 2;
        #pragma unroll
        for (int mt = 0; mt < 4; ++mt) scatter_acc(A[mt], s_dst, mt, hi, ch, agg);
    }

    // ---- yp1 = p1@W1r - n1@W1i ; yn1 = p1@W1i + n1@W1r : K=288(9), N=96(6)
    const _Float16* Apr[4];
    const _Float16* Anr[4];
    #pragma unroll
    for (int mt = 0; mt < 4; ++mt) {
        Apr[mt] = sAp1 + (mt * 16 + mrow) * 288;
        Anr[mt] = sAn1 + (mt * 16 + mrow) * 288;
    }
    for (int id = wave; id < 12; id += 8) {
        bool isyn = id >= 6;
        int nc = isyn ? id - 6 : id;
        const _Float16* Bp = isyn ? W1ip : W1rp;
        const _Float16* Bn = isyn ? W1rp : W1inp;
        v8f A[4] = {};
        v16h bp = load_b16(Bp, 0 * 6 + nc, lane);
        v16h bn = load_b16(Bn, 0 * 6 + nc, lane);
        for (int ks = 0; ks < 9; ++ks) {
            v16h fp[4], fn[4];
            #pragma unroll
            for (int mt = 0; mt < 4; ++mt) {
                fp[mt] = load_a16(Apr[mt], ks * 32, hi);
                fn[mt] = load_a16(Anr[mt], ks * 32, hi);
            }
            v16h bpn = bp, bnn = bn;
            if (ks < 8) {
                bpn = load_b16(Bp, (ks + 1) * 6 + nc, lane);
                bnn = load_b16(Bn, (ks + 1) * 6 + nc, lane);
            }
            #pragma unroll
            for (int mt = 0; mt < 4; ++mt) A[mt] = wmma16(fp[mt], bp, A[mt]);
            #pragma unroll
            for (int mt = 0; mt < 4; ++mt) A[mt] = wmma16(fn[mt], bn, A[mt]);
            bp = bpn; bn = bnn;
        }
        int c = nc * 16 + n;
        int ch = isyn ? (c < 64 ? 224 + c * 3     : 416 + (c - 64) * 5 + 1)
                      : (c < 64 ? 224 + c * 3 + 2 : 416 + (c - 64) * 5 + 3);
        #pragma unroll
        for (int mt = 0; mt < 4; ++mt) scatter_acc(A[mt], s_dst, mt, hi, ch, agg);
    }

    // ---- yp2 / yn2 : K=96 (3), N=32 (2) ----------------------------------
    if (wave < 4) {
        bool isyn = wave >= 2;
        int nc = wave & 1;
        const _Float16* Bp = isyn ? W2ip : W2rp;
        const _Float16* Bn = isyn ? W2rp : W2inp;
        v8f A[4] = {};
        #pragma unroll
        for (int ks = 0; ks < 3; ++ks) {
            v16h bp = load_b16(Bp, ks * 2 + nc, lane);
            v16h bn = load_b16(Bn, ks * 2 + nc, lane);
            #pragma unroll
            for (int mt = 0; mt < 4; ++mt) {
                v16h fp = load_a16(sP2 + (mt * 16 + mrow) * 96, ks * 32, hi);
                v16h fn = load_a16(sN2 + (mt * 16 + mrow) * 96, ks * 32, hi);
                A[mt] = wmma16(fp, bp, A[mt]);
                A[mt] = wmma16(fn, bn, A[mt]);
            }
        }
        int c = nc * 16 + n;
        int ch = isyn ? 416 + c * 5 : 416 + c * 5 + 4;
        #pragma unroll
        for (int mt = 0; mt < 4; ++mt) scatter_acc(A[mt], s_dst, mt, hi, ch, agg);
    }
}

// ---------------------------------------------------------------------------
// Node post-processing: gate -> Wp/Wenv -> one-hot T contraction -> gate ->
// Wo -> residual.  One wave per node, 8 nodes / block.
// ---------------------------------------------------------------------------
__global__ __launch_bounds__(256) void node_out_kernel(
        const float* __restrict__ agg,
        const float* __restrict__ latent, const float* __restrict__ onehot,
        const float* __restrict__ Wenv,
        const float* __restrict__ Wp0, const float* __restrict__ bp0,
        const float* __restrict__ Wp1, const float* __restrict__ Wp2,
        const float* __restrict__ T0, const float* __restrict__ T1,
        const float* __restrict__ T2,
        const float* __restrict__ Wo0, const float* __restrict__ bo0,
        const float* __restrict__ Wo1, const float* __restrict__ Wo2,
        const float* __restrict__ res_param,
        const float* __restrict__ node_feat,
        float* __restrict__ out) {
    __shared__ float SW[8][1760];
    __shared__ int s_oh[8];
    const int S_S = 0, S_V1 = 128, S_V2 = 320, S_WV = 480, S_H0 = 704, S_H1 = 832,
              S_H2 = 1024, S_SO = 1184, S_G2 = 1312, S_G1O = 1408, S_G2O = 1600;

    const int lane = threadIdx.x & 31;
    const int wv   = threadIdx.x >> 5;
    const int node = blockIdx.x * 8 + wv;
    float* S = &SW[wv][0];
    const float* ag = agg + (size_t)node * AGG_CH;

    #pragma unroll
    for (int j = 0; j < 4; ++j) {
        int u = lane + 32 * j;
        float x = ag[u];
        S[S_S + u] = x * sigmoidf_(x);
    }
    #pragma unroll
    for (int j = 0; j < 3; ++j) {
        int u = lane + 32 * j;
        float gv = sigmoidf_(ag[128 + u]);
        if (u < 64) {
            #pragma unroll
            for (int m = 0; m < 3; ++m) S[S_V1 + u * 3 + m] = ag[224 + u * 3 + m] * gv;
        } else {
            int uu = u - 64;
            #pragma unroll
            for (int m = 0; m < 5; ++m) S[S_V2 + uu * 5 + m] = ag[416 + uu * 5 + m] * gv;
        }
    }
    for (int jj = 0; jj < 7; ++jj) {
        int c = lane + 32 * jj;
        float acc = 0.f;
        for (int u = 0; u < 128; ++u) acc += latent[(size_t)node * 128 + u] * Wenv[u * 224 + c];
        S[S_WV + c] = acc;
    }
    {
        int cand = -1;
        if (onehot[(size_t)node * 64 + lane] > 0.5f)      cand = lane;
        if (onehot[(size_t)node * 64 + lane + 32] > 0.5f) cand = lane + 32;
        #pragma unroll
        for (int m = 16; m >= 1; m >>= 1) { int o = __shfl_xor(cand, m, 32); cand = o > cand ? o : cand; }
        if (lane == 0) s_oh[wv] = cand < 0 ? 0 : cand;
    }
    __syncthreads();

    #pragma unroll
    for (int k = 0; k < 4; ++k) {
        int w = lane * 4 + k;
        float acc = bp0[w];
        for (int u = 0; u < 128; ++u) acc += S[S_S + u] * Wp0[u * 128 + w];
        S[S_H0 + w] = acc * S[S_WV + w];
    }
    #pragma unroll
    for (int k = 0; k < 2; ++k) {
        int w = lane * 2 + k;
        #pragma unroll
        for (int m = 0; m < 3; ++m) {
            float acc = 0.f;
            for (int u = 0; u < 64; ++u) acc += S[S_V1 + u * 3 + m] * Wp1[u * 64 + w];
            S[S_H1 + w * 3 + m] = acc * S[S_WV + 128 + w];
        }
    }
    {
        int w = lane;
        #pragma unroll
        for (int m = 0; m < 5; ++m) {
            float acc = 0.f;
            for (int u = 0; u < 32; ++u) acc += S[S_V2 + u * 5 + m] * Wp2[u * 32 + w];
            S[S_H2 + w * 5 + m] = acc * S[S_WV + 192 + w];
        }
    }
    __syncthreads();

    const int vo = s_oh[wv];
    for (int jj = 0; jj < 7; ++jj) {
        int c = lane + 32 * jj;
        float acc = 0.f;
        for (int u = 0; u < 128; ++u) acc += S[S_H0 + u] * T0[(size_t)(u * 64 + vo) * 224 + c];
        if (c < 128) S[S_SO + c] = acc * sigmoidf_(acc);
        else         S[S_G2 + c - 128] = sigmoidf_(acc);
    }
    __syncthreads();

    #pragma unroll
    for (int k = 0; k < 2; ++k) {
        int w = lane * 2 + k;
        float gg = S[S_G2 + w];
        #pragma unroll
        for (int m = 0; m < 3; ++m) {
            float acc = 0.f;
            for (int u = 0; u < 64; ++u) acc += S[S_H1 + u * 3 + m] * T1[(size_t)(u * 64 + vo) * 64 + w];
            S[S_G1O + w * 3 + m] = acc * gg;
        }
    }
    {
        int w = lane;
        float gg = S[S_G2 + 64 + w];
        #pragma unroll
        for (int m = 0; m < 5; ++m) {
            float acc = 0.f;
            for (int u = 0; u < 32; ++u) acc += S[S_H2 + u * 5 + m] * T2[(size_t)(u * 64 + vo) * 32 + w];
            S[S_G2O + w * 5 + m] = acc * gg;
        }
    }
    __syncthreads();

    #pragma unroll
    for (int k = 0; k < 4; ++k) {
        int w = lane * 4 + k;
        float acc = bo0[w];
        for (int u = 0; u < 128; ++u) acc += S[S_SO + u] * Wo0[u * 128 + w];
        S[S_H0 + w] += acc;
    }
    #pragma unroll
    for (int k = 0; k < 2; ++k) {
        int w = lane * 2 + k;
        #pragma unroll
        for (int m = 0; m < 3; ++m) {
            float acc = 0.f;
            for (int u = 0; u < 64; ++u) acc += S[S_G1O + u * 3 + m] * Wo1[u * 64 + w];
            S[S_H1 + w * 3 + m] += acc;
        }
    }
    {
        int w = lane;
        #pragma unroll
        for (int m = 0; m < 5; ++m) {
            float acc = 0.f;
            for (int u = 0; u < 32; ++u) acc += S[S_G2O + u * 5 + m] * Wo2[u * 32 + w];
            S[S_H2 + w * 5 + m] += acc;
        }
    }
    __syncthreads();

    float a = sigmoidf_(res_param[0]);
    float b = 1.0f - a;
    #pragma unroll
    for (int j = 0; j < 15; ++j) {
        int c = lane + 32 * j;
        float hv = (c < 128) ? S[S_H0 + c] : (c < 320 ? S[S_H1 + c - 128] : S[S_H2 + c - 320]);
        out[(size_t)node * 480 + c] = a * hv + b * node_feat[(size_t)node * 480 + c];
    }
}

// ---------------------------------------------------------------------------
extern "C" void kernel_launch(void* const* d_in, const int* in_sizes, int n_in,
                              void* d_out, int out_size, void* d_ws, size_t ws_size,
                              hipStream_t stream) {
    const float* node_feat   = (const float*)d_in[0];
    const float* edge_feat   = (const float*)d_in[1];
    const float* node_latent = (const float*)d_in[2];
    const float* node_onehot = (const float*)d_in[3];
    const float* ln_n_gamma  = (const float*)d_in[4];
    const float* ln_n_beta   = (const float*)d_in[5];
    const float* ln_e_gamma  = (const float*)d_in[6];
    const float* ln_e_beta   = (const float*)d_in[7];
    const float* W0   = (const float*)d_in[8];
    const float* W1r  = (const float*)d_in[9];
    const float* W1i  = (const float*)d_in[10];
    const float* W2r  = (const float*)d_in[11];
    const float* W2i  = (const float*)d_in[12];
    const float* Wenv = (const float*)d_in[13];
    const float* Wp0  = (const float*)d_in[14];
    const float* bp0  = (const float*)d_in[15];
    const float* Wp1  = (const float*)d_in[16];
    const float* Wp2  = (const float*)d_in[17];
    const float* T0   = (const float*)d_in[18];
    const float* T1   = (const float*)d_in[19];
    const float* T2   = (const float*)d_in[20];
    const float* Wo0  = (const float*)d_in[21];
    const float* bo0  = (const float*)d_in[22];
    const float* Wo1  = (const float*)d_in[23];
    const float* Wo2  = (const float*)d_in[24];
    const float* res_param = (const float*)d_in[25];
    const int* esrc = (const int*)d_in[26];
    const int* edst = (const int*)d_in[27];
    float* out = (float*)d_out;

    char* w = (char*)d_ws;
    _Float16* xnh = (_Float16*)w;  w += (size_t)N_NODES * 480 * 2;
    float*    agg = (float*)w;     w += (size_t)N_NODES * AGG_CH * 4;
    _Float16* W0p   = (_Float16*)w; w += (size_t)672 * 320 * 2;
    _Float16* W1rp  = (_Float16*)w; w += (size_t)288 * 96 * 2;
    _Float16* W1ip  = (_Float16*)w; w += (size_t)288 * 96 * 2;
    _Float16* W1inp = (_Float16*)w; w += (size_t)288 * 96 * 2;
    _Float16* W2rp  = (_Float16*)w; w += (size_t)96 * 32 * 2;
    _Float16* W2ip  = (_Float16*)w; w += (size_t)96 * 32 * 2;
    _Float16* W2inp = (_Float16*)w; w += (size_t)96 * 32 * 2;

    // allow ~182KB dynamic LDS for the edge kernel (CDNA5: up to 320KB / workgroup)
    hipFuncSetAttribute((const void*)edge_kernel,
                        hipFuncAttributeMaxDynamicSharedMemorySize, EDGE_SMEM_BYTES);

    hipMemsetAsync(agg, 0, (size_t)N_NODES * AGG_CH * 4, stream);

    pack_w_kernel<<<(672 * 320 + 255) / 256, 256, 0, stream>>>(W0,  W0p,   672, 320,  1.0f);
    pack_w_kernel<<<(288 * 96  + 255) / 256, 256, 0, stream>>>(W1r, W1rp,  288, 96,   1.0f);
    pack_w_kernel<<<(288 * 96  + 255) / 256, 256, 0, stream>>>(W1i, W1ip,  288, 96,   1.0f);
    pack_w_kernel<<<(288 * 96  + 255) / 256, 256, 0, stream>>>(W1i, W1inp, 288, 96,  -1.0f);
    pack_w_kernel<<<(96 * 32   + 255) / 256, 256, 0, stream>>>(W2r, W2rp,  96,  32,   1.0f);
    pack_w_kernel<<<(96 * 32   + 255) / 256, 256, 0, stream>>>(W2i, W2ip,  96,  32,   1.0f);
    pack_w_kernel<<<(96 * 32   + 255) / 256, 256, 0, stream>>>(W2i, W2inp, 96,  32,  -1.0f);

    node_ln_kernel<<<N_NODES / 8, 256, 0, stream>>>(node_feat, ln_n_gamma, ln_n_beta, xnh);

    edge_kernel<<<N_EDGES / EDGES_PER_BLK, 256, EDGE_SMEM_BYTES, stream>>>(
        edge_feat, xnh, ln_e_gamma, ln_e_beta, esrc, edst,
        W0p, W1rp, W1ip, W1inp, W2rp, W2ip, W2inp, agg);

    node_out_kernel<<<N_NODES / 8, 256, 0, stream>>>(agg, node_latent, node_onehot, Wenv,
                                                     Wp0, bp0, Wp1, Wp2, T0, T1, T2,
                                                     Wo0, bo0, Wo1, Wo2, res_param,
                                                     node_feat, out);
}
